// ManifoldLayer_78795470012861
// MI455X (gfx1250) — matrix operand order repaired and checked by reading, not compiled, not run on hardware
//
#include <hip/hip_runtime.h>
#include <math.h>

// Problem constants (from reference): B=8, S=4096, H=8, D=64
#define BB   8
#define SS   4096
#define HH   8
#define DD   64
#define HDD  512                      // H*D
#define BF   (BB*SS)                  // 32768 rows
#define NEL  ((size_t)BF*(size_t)HDD) // 16,777,216 elements per [Bf,HD] matrix
#define WSZ  (512*512)                // one weight matrix element count

typedef __attribute__((ext_vector_type(16))) __bf16 v16bf;
typedef __attribute__((ext_vector_type(8)))  __bf16 v8bf;
typedef __attribute__((ext_vector_type(8)))  float  v8f;

__device__ __forceinline__ float wrapf(float a) {
  const float PI_F   = 3.14159265358979323846f;
  const float TWO_PI = 6.28318530717958647692f;
  float t = (a + PI_F) * (1.0f / TWO_PI);
  t -= floorf(t);                 // python-style mod -> [0,1)
  return t * TWO_PI - PI_F;
}
__device__ __forceinline__ float sigm(float x) { return 1.0f / (1.0f + expf(-x)); }

// ---------------------------------------------------------------------------
// Kernel 1: weights f32 [K][N] -> bf16 transposed [mat][N][K]
// ---------------------------------------------------------------------------
__global__ __launch_bounds__(256) void prep_weights(
    const float* __restrict__ Wx, const float* __restrict__ Wv,
    const float* __restrict__ Wdx, const float* __restrict__ Wdv,
    __bf16* __restrict__ Wt)
{
  int idx = blockIdx.x * 256 + threadIdx.x;   // 0 .. 4*512*512-1
  int mat = idx >> 18;
  int r   = idx & ((1 << 18) - 1);
  int n   = r & 511;        // consecutive threads -> consecutive n (coalesced read)
  int k   = r >> 9;
  const float* W = (mat == 0) ? Wx : (mat == 1) ? Wv : (mat == 2) ? Wdx : Wdv;
  Wt[(size_t)mat * WSZ + (size_t)n * HDD + k] = (__bf16)W[(size_t)k * HDD + n];
}

// ---------------------------------------------------------------------------
// Kernel 2: per-(row,head) gate + symplectic Euler + torus wrap -> bf16 A mats
// One wave per (row,head); lane handles elements d=lane and d=lane+32.
// ---------------------------------------------------------------------------
__global__ __launch_bounds__(256) void prep_rows(
    const float* __restrict__ x, const float* __restrict__ v,
    const float* __restrict__ f, const float* __restrict__ dtp,
    const float* __restrict__ Wg, const float* __restrict__ bg,
    __bf16* __restrict__ XsB, __bf16* __restrict__ VsB, __bf16* __restrict__ CxB)
{
  int lane = threadIdx.x & 31;
  int wid  = blockIdx.x * 8 + (threadIdx.x >> 5);   // 0 .. Bf*H-1
  int h    = wid & (HH - 1);
  size_t base = (size_t)wid * DD;

  float x0 = x[base + lane], x1 = x[base + lane + 32];
  float v0 = v[base + lane], v1 = v[base + lane + 32];
  float f0 = f[base + lane], f1 = f[base + lane + 32];

  const float* wg = Wg + (size_t)h * (2 * DD);
  float p = cosf(x0) * wg[lane]      + sinf(x0) * wg[DD + lane]
          + cosf(x1) * wg[lane + 32] + sinf(x1) * wg[DD + lane + 32];
  // wave32 all-reduce
  #pragma unroll
  for (int m = 16; m >= 1; m >>= 1) p += __shfl_xor(p, m, 32);

  float dp = dtp[h];
  float sp = (dp > 20.0f) ? dp : log1pf(expf(dp));      // softplus
  sp = fminf(fmaxf(sp, 1e-3f), 1.0f);                   // clamp
  float dt = sp * sigm(p + bg[h]);

  float vs0 = v0 + dt * f0, vs1 = v1 + dt * f1;
  float xs0 = wrapf(x0 + dt * vs0), xs1 = wrapf(x1 + dt * vs1);

  XsB[base + lane] = (__bf16)xs0;  XsB[base + lane + 32] = (__bf16)xs1;
  VsB[base + lane] = (__bf16)vs0;  VsB[base + lane + 32] = (__bf16)vs1;
  CxB[base + lane] = (__bf16)x0;   CxB[base + lane + 32] = (__bf16)x1;
}

// A fragment loader: 16-bit A 16x32 layout (ISA 7.12.2):
// lane group kg=lane>>4; halfs 0..7 -> k = 8*kg+0..7, halfs 8..15 -> k = 16+8*kg+0..7
__device__ __forceinline__ v16bf loadA(const __bf16* p) {
  v8bf lo = *reinterpret_cast<const v8bf*>(p);
  v8bf hi = *reinterpret_cast<const v8bf*>(p + 16);
  return __builtin_shufflevector(lo, hi, 0,1,2,3,4,5,6,7,8,9,10,11,12,13,14,15);
}
// B fragment loader: 32x16 B, same k order as A:
// halfs 0..7 -> k = 8*kg+0..7, halfs 8..15 -> k = 16+8*kg+0..7
__device__ __forceinline__ v16bf loadB(const __bf16* p) {
  v8bf lo = *reinterpret_cast<const v8bf*>(p);
  v8bf hi = *reinterpret_cast<const v8bf*>(p + 16);
  return __builtin_shufflevector(lo, hi, 0,1,2,3,4,5,6,7,8,9,10,11,12,13,14,15);
}

#define WMMA_BF16(A, Bm, C) \
  __builtin_amdgcn_wmma_f32_16x16x32_bf16(false, (A), false, (Bm), (short)0, (C), false, false)

// ---------------------------------------------------------------------------
// Kernel 3: fused 4-way GEMM (bf16 WMMA, f32 acc) + full epilogue.
// Block = 512 thr (16 waves) owns 32 rows x all 512 cols.
// Wave w: M-half mh=w>>3 (16 rows), N-slice (w&7)*64 (4 N-tiles).
// Halves L2 weight traffic vs 16-row blocks; wave pairs share B frags via L0.
// ---------------------------------------------------------------------------
__global__ __launch_bounds__(512) void fused_gemm(
    const __bf16* __restrict__ XsB, const __bf16* __restrict__ VsB,
    const __bf16* __restrict__ CxB, const __bf16* __restrict__ Wt,
    const float* __restrict__ bx,  const float* __restrict__ bv,
    const float* __restrict__ bdx, const float* __restrict__ bdv,
    const float* __restrict__ x,   const float* __restrict__ v,
    float* __restrict__ outx, float* __restrict__ outv)
{
  int lane = threadIdx.x & 31;
  int wave = threadIdx.x >> 5;   // 0..15
  int mh   = wave >> 3;          // row half: 0 or 1
  int w8   = wave & 7;           // column slice
  int mrow = lane & 15;          // A: row m / B: col n / C/D: col n
  int kg   = lane >> 4;          // lane half
  int row0 = blockIdx.x * 32 + mh * 16;

  const __bf16* WtX  = Wt;
  const __bf16* WtV  = Wt + (size_t)1 * WSZ;
  const __bf16* WtDX = Wt + (size_t)2 * WSZ;
  const __bf16* WtDV = Wt + (size_t)3 * WSZ;

  v8f accX[4], accV[4], accGX[4], accGV[4];
  v8f z = {0.f, 0.f, 0.f, 0.f, 0.f, 0.f, 0.f, 0.f};
  #pragma unroll
  for (int t = 0; t < 4; ++t) { accX[t] = z; accV[t] = z; accGX[t] = z; accGV[t] = z; }

  size_t arow = (size_t)(row0 + mrow) * HDD + 8 * kg;   // A frag base
  const __bf16* aXp = XsB + arow;
  const __bf16* aVp = VsB + arow;
  const __bf16* aCp = CxB + arow;
  size_t brow = (size_t)(w8 * 64 + mrow) * HDD + 8 * kg;  // B frag base (tile nt adds 16 rows)

  for (int kb = 0; kb < HDD; kb += 32) {
    v16bf aX = loadA(aXp + kb);
    v16bf aV = loadA(aVp + kb);
    v16bf aC = loadA(aCp + kb);
    #pragma unroll
    for (int nt = 0; nt < 4; ++nt) {
      size_t bo = brow + (size_t)nt * 16 * HDD + kb;
      v16bf b0 = loadB(WtX  + bo);  accX[nt]  = WMMA_BF16(aX, b0, accX[nt]);
      v16bf b1 = loadB(WtV  + bo);  accV[nt]  = WMMA_BF16(aV, b1, accV[nt]);
      v16bf b2 = loadB(WtDX + bo);  accGX[nt] = WMMA_BF16(aC, b2, accGX[nt]);
      v16bf b3 = loadB(WtDV + bo);  accGV[nt] = WMMA_BF16(aC, b3, accGV[nt]);
    }
  }

  // Epilogue. C/D layout: vgpr i, lane -> element (m = 8*kg + i, n = mrow).
  float vkeep[4][8];
  float sum2[8];
  #pragma unroll
  for (int i = 0; i < 8; ++i) sum2[i] = 0.f;

  #pragma unroll
  for (int nt = 0; nt < 4; ++nt) {
    int n = w8 * 64 + nt * 16 + mrow;
    float bxn = bx[n], bvn = bv[n], bdxn = bdx[n], bdvn = bdv[n];
    #pragma unroll
    for (int i = 0; i < 8; ++i) {
      int r = row0 + 8 * kg + i;
      size_t off = (size_t)r * HDD + n;
      float xm  = tanhf(accX[nt][i] + bxn);
      float vm  = tanhf(accV[nt][i] + bvn);
      float g1  = sigm(accGX[nt][i] + bdxn);
      float g2  = sigm(accGV[nt][i] + bdvn);
      float cx  = x[off];
      float cv  = v[off];
      outx[off] = wrapf(cx + g1 * wrapf(xm - cx));
      float vn  = cv + g2 * (vm - cv);
      vkeep[nt][i] = vn;
      sum2[i] += vn * vn;
    }
  }

  // Reduce sum2 across the 16 lanes of each lane-half (same rows, diff cols)
  #pragma unroll
  for (int m = 8; m >= 1; m >>= 1) {
    #pragma unroll
    for (int i = 0; i < 8; ++i) sum2[i] += __shfl_xor(sum2[i], m, 32);
  }

  // Cross-wave reduction in LDS: red[wave][local row 0..15].
  // Row m_local (within the wave's 16-row half) is covered by the 8 waves
  // sharing the same mh.
  __shared__ float red[16][16];
  if ((lane & 15) == 0) {
    #pragma unroll
    for (int i = 0; i < 8; ++i) red[wave][8 * kg + i] = sum2[i];
  }
  __syncthreads();

  float scale[8];
  #pragma unroll
  for (int i = 0; i < 8; ++i) {
    int m = 8 * kg + i;          // local row within this wave's half
    float t = 0.f;
    #pragma unroll
    for (int w2 = 0; w2 < 8; ++w2) t += red[mh * 8 + w2][m];
    scale[i] = rsqrtf(t * (1.0f / (float)HDD) + 1e-6f);   // rsqrt(mean + eps)
  }

  #pragma unroll
  for (int nt = 0; nt < 4; ++nt) {
    int n = w8 * 64 + nt * 16 + mrow;
    #pragma unroll
    for (int i = 0; i < 8; ++i) {
      int r = row0 + 8 * kg + i;
      outv[(size_t)r * HDD + n] = vkeep[nt][i] * scale[i];
    }
  }
}

// ---------------------------------------------------------------------------
// Launcher. Inputs in setup_inputs() order:
// 0:x 1:v 2:force 3:dt_params 4:Wg 5:bg 6:Wx 7:bx 8:Wv 9:bv 10:Wdx 11:bdx 12:Wdv 13:bdv
// d_out = [x_next (NEL f32) | v_next (NEL f32)]
// ---------------------------------------------------------------------------
extern "C" void kernel_launch(void* const* d_in, const int* in_sizes, int n_in,
                              void* d_out, int out_size, void* d_ws, size_t ws_size,
                              hipStream_t stream) {
  const float* x   = (const float*)d_in[0];
  const float* v   = (const float*)d_in[1];
  const float* f   = (const float*)d_in[2];
  const float* dtp = (const float*)d_in[3];
  const float* Wg  = (const float*)d_in[4];
  const float* bg  = (const float*)d_in[5];
  const float* Wx  = (const float*)d_in[6];
  const float* bx  = (const float*)d_in[7];
  const float* Wv  = (const float*)d_in[8];
  const float* bv  = (const float*)d_in[9];
  const float* Wdx = (const float*)d_in[10];
  const float* bdx = (const float*)d_in[11];
  const float* Wdv = (const float*)d_in[12];
  const float* bdv = (const float*)d_in[13];

  float* outx = (float*)d_out;
  float* outv = outx + NEL;

  // Workspace carve-up (bf16): Xs | Vs | Cx | Wt[4 matrices, transposed]
  __bf16* XsB = (__bf16*)d_ws;
  __bf16* VsB = XsB + NEL;
  __bf16* CxB = VsB + NEL;
  __bf16* Wt  = CxB + NEL;   // 4 * 512*512 bf16

  prep_weights<<<(4 * WSZ) / 256, 256, 0, stream>>>(Wx, Wv, Wdx, Wdv, Wt);
  prep_rows<<<(BF * HH) / 8, 256, 0, stream>>>(x, v, f, dtp, Wg, bg, XsB, VsB, CxB);
  fused_gemm<<<BF / 32, 512, 0, stream>>>(XsB, VsB, CxB, Wt,
                                          bx, bv, bdx, bdv, x, v, outx, outv);
}
